// GCN_67843303408207
// MI455X (gfx1250) — matrix operand order repaired and checked by reading, not compiled, not run on hardware
//
#include <hip/hip_runtime.h>
#include <hip/hip_bf16.h>
#include <stdint.h>

// ---------------------------------------------------------------------------
// Types for CDNA5 WMMA (wave32): A/B = 16 bf16 per lane, C/D = 8 f32 per lane
// ---------------------------------------------------------------------------
typedef __attribute__((ext_vector_type(16))) __bf16        v16bf;
typedef __attribute__((ext_vector_type(2)))  __bf16        v2bf;
typedef __attribute__((ext_vector_type(8)))  unsigned int  v8u;
typedef __attribute__((ext_vector_type(8)))  float         v8f;

__device__ __forceinline__ unsigned int pack2_bf16(float lo, float hi) {
    // lowers to a single v_cvt_pk_bf16_f32: low half = even K, high half = odd K
    v2bf p;
    p.x = (__bf16)lo;
    p.y = (__bf16)hi;
    return __builtin_bit_cast(unsigned int, p);
}

// ---------------------------------------------------------------------------
// Pre-pack a weight matrix W[KTOT x NCOLS] (f32, row-major) into K-pair-
// interleaved bf16 words: WP[(k/2)*NCOLS + col] = {bf16 W[k][col], W[k+1][col]},
// zero-padded up to KPAD (multiple of 32). This makes every WMMA B fragment a
// straight 8x b32 load + bit_cast (no conversion VALU in the GEMM loop).
// ---------------------------------------------------------------------------
__global__ void gcn_pack_w(const float* __restrict__ W, unsigned int* __restrict__ WP,
                           int KTOT, int KPAD, int NCOLS)
{
    int idx = blockIdx.x * blockDim.x + threadIdx.x;
    int total = (KPAD / 2) * NCOLS;
    if (idx >= total) return;
    int wrow = idx / NCOLS;
    int col  = idx - wrow * NCOLS;
    int k0 = 2 * wrow;
    float f0 = (k0     < KTOT) ? W[(size_t)k0 * NCOLS + col]       : 0.0f;
    float f1 = (k0 + 1 < KTOT) ? W[(size_t)(k0 + 1) * NCOLS + col] : 0.0f;
    WP[idx] = pack2_bf16(f0, f1);
}

// ---------------------------------------------------------------------------
// Dense GEMM: C[nrows x NCOLS] = act(A[nrows x LDA](+biasPrev)) @ W[KTOT x NCOLS]
// Each wave computes a 16-row strip.  A tile: float2 coalesced loads ->
// bias/ReLU -> packed bf16 pairs in LDS (word stride 17, conflict-free) ->
// 8x ds_load_b32 per fragment.  B: 8x b32 loads from pre-packed bf16.
// Accumulate with v_wmma_f32_16x16x32_bf16 (fp32 accumulate).
// ---------------------------------------------------------------------------
template<int LDA, int KTOT, int NCOLS, bool BIASRELU>
__global__ __launch_bounds__(128) void gcn_gemm_wmma(
    const float* __restrict__ A, const unsigned int* __restrict__ WP,
    const float* __restrict__ biasPrev, float* __restrict__ C, int nrows)
{
    constexpr int NT     = NCOLS / 16;         // 16-wide column tiles
    constexpr int KSTEPS = (KTOT + 31) / 32;   // K padded to 32

    __shared__ unsigned int sA[4][16 * 17];    // packed bf16 pairs, stride 17

    const int wave = threadIdx.x >> 5;
    const int lane = threadIdx.x & 31;
    const int rowTile = blockIdx.x * 4 + wave;
    if (rowTile * 16 >= nrows) return;         // wave-uniform exit: EXEC stays all-1s
    const int rowBase = rowTile * 16;

    unsigned int* sm = sA[wave];
    const int m = lane & 15;                   // fragment row / column within tile
    const int h = lane >> 4;                   // lane half selects K sub-range
    const int n = lane & 15;

    v8f acc[NT];
    #pragma unroll
    for (int t = 0; t < NT; ++t) acc[t] = v8f{0.f,0.f,0.f,0.f,0.f,0.f,0.f,0.f};

    #pragma unroll
    for (int ks = 0; ks < KSTEPS; ++ks) {
        const int kbase = ks * 32;

        // ---- stage 16x32 A tile into LDS as packed bf16 pairs (coalesced) ----
        #pragma unroll
        for (int i = 0; i < 8; ++i) {
            int idx = i * 32 + lane;           // 0..255 packed words
            int r = idx >> 4;                  // row 0..15
            int w = idx & 15;                  // word 0..15  (K = 2w, 2w+1)
            int k = kbase + 2 * w;
            float f0, f1;
            if constexpr (KTOT % 32 == 0) {
                const float2 v2 = *(const float2*)&A[(size_t)(rowBase + r) * LDA + k];
                f0 = v2.x; f1 = v2.y;
                if constexpr (BIASRELU) {
                    f0 += biasPrev[k];     f0 = f0 > 0.0f ? f0 : 0.0f;
                    f1 += biasPrev[k + 1]; f1 = f1 > 0.0f ? f1 : 0.0f;
                }
            } else {                            // guarded tail (K=48, ks=1)
                f0 = 0.0f; f1 = 0.0f;
                if (k < KTOT) {
                    f0 = A[(size_t)(rowBase + r) * LDA + k];
                    if constexpr (BIASRELU) { f0 += biasPrev[k]; f0 = f0 > 0.0f ? f0 : 0.0f; }
                }
                if (k + 1 < KTOT) {
                    f1 = A[(size_t)(rowBase + r) * LDA + k + 1];
                    if constexpr (BIASRELU) { f1 += biasPrev[k + 1]; f1 = f1 > 0.0f ? f1 : 0.0f; }
                }
            }
            sm[r * 17 + w] = pack2_bf16(f0, f1);
        }
        if (ks + 1 < KSTEPS)
            __builtin_prefetch(&A[(size_t)rowBase * LDA + kbase + 32], 0, 1);

        // ---- A fragment: 16x32 A in contiguous K = 16h + s order, from packed LDS words ----
        // VGPR j -> K pair word (8h + j), i.e. K = 16h + s (matches B)
        v8u aw;
        #pragma unroll
        for (int j = 0; j < 4; ++j) aw[j]     = sm[m * 17 + 8 * h + j];
        #pragma unroll
        for (int j = 0; j < 4; ++j) aw[4 + j] = sm[m * 17 + 8 * h + 4 + j];
        v16bf a = __builtin_bit_cast(v16bf, aw);

        // ---- B fragments from pre-packed weights ----
        // lanes 0-15 hold K=kbase..+15, lanes 16-31 +16; word row = k/2
        const int kwb = kbase / 2 + 8 * h;
        #pragma unroll
        for (int t = 0; t < NT; ++t) {
            v8u bw;
            #pragma unroll
            for (int j = 0; j < 8; ++j)
                bw[j] = WP[(size_t)(kwb + j) * NCOLS + t * 16 + n];
            v16bf b = __builtin_bit_cast(v16bf, bw);
            acc[t] = __builtin_amdgcn_wmma_f32_16x16x32_bf16(
                         false, a, false, b, (short)0, acc[t], false, false);
        }
    }

    // ---- store C: VGPR r holds (M = 8h + r, N = lane&15) ----
    #pragma unroll
    for (int t = 0; t < NT; ++t) {
        #pragma unroll
        for (int r = 0; r < 8; ++r) {
            int mrow = rowBase + 8 * h + r;
            if (mrow < nrows)
                C[(size_t)mrow * NCOLS + t * 16 + n] = acc[t][r];
        }
    }
}

// ---------------------------------------------------------------------------
// COO SpMM: out[row[e]] += val[e] * dense[col[e]]   (edge-parallel, F lanes/edge)
// All targets are L2-resident (<=25.6MB), so f32 atomics are cheap.
// ---------------------------------------------------------------------------
template<int F>
__global__ __launch_bounds__(256) void gcn_spmm_atomic(
    const int* __restrict__ row, const int* __restrict__ col,
    const float* __restrict__ val, const float* __restrict__ dense,
    float* __restrict__ out, int E)
{
    constexpr int GP  = (F + 31) & ~31;   // lanes per edge (32 or 64)
    constexpr int EPB = 256 / GP;         // edges per block
    const int e = blockIdx.x * EPB + (int)(threadIdx.x / GP);
    const int f = (int)(threadIdx.x % GP);
    if (e >= E || f >= F) return;
    const int r = row[e];
    const int c = col[e];
    const float v = val[e];
    atomicAdd(&out[(size_t)r * F + f], v * dense[(size_t)c * F + f]);
}

__global__ void gcn_fill_zero(float* __restrict__ p, size_t n) {
    size_t i = (size_t)blockIdx.x * blockDim.x + threadIdx.x;
    if (i < n) p[i] = 0.0f;
}

// ---------------------------------------------------------------------------
// Global mean pool over nodes of relu(h + b3): partial sums per block, one
// atomic per column per block.  blockDim = 64 (one lane per feature).
// ---------------------------------------------------------------------------
__global__ __launch_bounds__(64) void gcn_pool64(
    const float* __restrict__ h, const float* __restrict__ b3,
    float* __restrict__ pooled, int nrows)
{
    const int c = threadIdx.x;
    const int chunk = (nrows + gridDim.x - 1) / gridDim.x;
    const int r0 = blockIdx.x * chunk;
    int r1 = r0 + chunk; if (r1 > nrows) r1 = nrows;
    const float bb = b3[c];
    float s = 0.0f;
    for (int r = r0; r < r1; ++r) {
        float v = h[(size_t)r * 64 + c] + bb;
        s += (v > 0.0f) ? v : 0.0f;
    }
    atomicAdd(&pooled[c], s);
}

// ---------------------------------------------------------------------------
// Head: y = mean-pooled[64]; z = relu(y@fc1W+fc1b); out = softmax(z@fc2W+fc2b)
// ---------------------------------------------------------------------------
__global__ __launch_bounds__(64) void gcn_head(
    const float* __restrict__ pooled,
    const float* __restrict__ fc1W, const float* __restrict__ fc1b,
    const float* __restrict__ fc2W, const float* __restrict__ fc2b,
    float* __restrict__ out, float invN)
{
    __shared__ float y[64];
    __shared__ float z[32];
    const int t = threadIdx.x;
    y[t] = pooled[t] * invN;
    __syncthreads();
    if (t < 32) {
        float s = fc1b[t];
        #pragma unroll 8
        for (int c = 0; c < 64; ++c) s += y[c] * fc1W[c * 32 + t];
        z[t] = (s > 0.0f) ? s : 0.0f;
    }
    __syncthreads();
    if (t == 0) {
        float l0 = fc2b[0], l1 = fc2b[1];
        #pragma unroll 8
        for (int j = 0; j < 32; ++j) {
            l0 += z[j] * fc2W[j * 2 + 0];
            l1 += z[j] * fc2W[j * 2 + 1];
        }
        float mx = l0 > l1 ? l0 : l1;
        float e0 = __expf(l0 - mx), e1 = __expf(l1 - mx);
        float inv = 1.0f / (e0 + e1);
        out[0] = e0 * inv;
        out[1] = e1 * inv;
    }
}

// ---------------------------------------------------------------------------
extern "C" void kernel_launch(void* const* d_in, const int* in_sizes, int n_in,
                              void* d_out, int out_size, void* d_ws, size_t ws_size,
                              hipStream_t stream)
{
    const float* x    = (const float*)d_in[0];
    const int*   row  = (const int*)  d_in[1];
    const int*   col  = (const int*)  d_in[2];
    const float* val  = (const float*)d_in[3];
    const float* W1   = (const float*)d_in[4];
    const float* b1   = (const float*)d_in[5];
    const float* W2   = (const float*)d_in[6];
    const float* b2   = (const float*)d_in[7];
    const float* W3   = (const float*)d_in[8];
    const float* b3   = (const float*)d_in[9];
    const float* fc1W = (const float*)d_in[10];
    const float* fc1b = (const float*)d_in[11];
    const float* fc2W = (const float*)d_in[12];
    const float* fc2b = (const float*)d_in[13];
    float* out = (float*)d_out;

    const int N = in_sizes[0] / 256;   // 100000
    const int E = in_sizes[1];         // 1600000

    // workspace: two ping-pong buffers [N x 64], pooled[64], packed weights
    float* buf0   = (float*)d_ws;
    float* buf1   = buf0 + (size_t)N * 64;
    float* pooled = buf1 + (size_t)N * 64;
    unsigned int* WP1 = (unsigned int*)(pooled + 64); // (256/2)*32 = 4096 words
    unsigned int* WP2 = WP1 + 128 * 32;               // (32/2)*48  =  768 words
    unsigned int* WP3 = WP2 + 16 * 48;                // (64/2)*64  = 2048 words

    const int rowTiles   = (N + 15) / 16;
    const int gemmBlocks = (rowTiles + 3) / 4;   // 4 waves / block

    // ---- pre-pack weights to bf16 K-pair-interleaved (tiny, L2-resident) ----
    gcn_pack_w<<<(128 * 32 + 255) / 256, 256, 0, stream>>>(W1, WP1, 256, 256, 32);
    gcn_pack_w<<<(16 * 48 + 255) / 256, 256, 0, stream>>>(W2, WP2, 32, 32, 48);
    gcn_pack_w<<<(32 * 64 + 255) / 256, 256, 0, stream>>>(W3, WP3, 48, 64, 64);

    // ---- layer 1: support = x @ W1 ; h1 = spmm(support) ----
    gcn_gemm_wmma<256, 256, 32, false><<<gemmBlocks, 128, 0, stream>>>(x, WP1, nullptr, buf0, N);
    {
        size_t n = (size_t)N * 32;
        gcn_fill_zero<<<(unsigned)((n + 255) / 256), 256, 0, stream>>>(buf1, n);
    }
    gcn_spmm_atomic<32><<<(E + 7) / 8, 256, 0, stream>>>(row, col, val, buf0, buf1, E);

    // ---- layer 2: support = relu(h1+b1) @ W2 ; h2 = spmm(support) ----
    gcn_gemm_wmma<32, 32, 48, true><<<gemmBlocks, 128, 0, stream>>>(buf1, WP2, b1, buf0, N);
    {
        size_t n = (size_t)N * 48;
        gcn_fill_zero<<<(unsigned)((n + 255) / 256), 256, 0, stream>>>(buf1, n);
    }
    gcn_spmm_atomic<48><<<(E + 3) / 4, 256, 0, stream>>>(row, col, val, buf0, buf1, E);

    // ---- layer 3: support = relu(h2+b2) @ W3 ; h3 = spmm(support) ----
    gcn_gemm_wmma<48, 48, 64, true><<<gemmBlocks, 128, 0, stream>>>(buf1, WP3, b2, buf0, N);
    {
        size_t n = (size_t)N * 64;
        gcn_fill_zero<<<(unsigned)((n + 255) / 256), 256, 0, stream>>>(buf1, n);
    }
    gcn_spmm_atomic<64><<<(E + 3) / 4, 256, 0, stream>>>(row, col, val, buf0, buf1, E);

    // ---- mean pool of relu(h3+b3), then MLP head + softmax ----
    gcn_fill_zero<<<1, 64, 0, stream>>>(pooled, 64);
    gcn_pool64<<<256, 64, 0, stream>>>(buf1, b3, pooled, N);
    gcn_head<<<1, 64, 0, stream>>>(pooled, fc1W, fc1b, fc2W, fc2b, out, 1.0f / (float)N);
}
